// MathFormulaExtractor_84619445666211
// MI455X (gfx1250) — matrix operand order repaired and checked by reading, not compiled, not run on hardware
//
#include <hip/hip_runtime.h>

// ---------------------------------------------------------------------------
// Types for CDNA5 WMMA (gfx1250, wave32)
// ---------------------------------------------------------------------------
typedef __attribute__((ext_vector_type(16))) __bf16 v16bf;
typedef __attribute__((ext_vector_type(8)))  float  v8f;

__device__ __forceinline__ unsigned short f2b(float f) {
    unsigned int u = __float_as_uint(f);
    unsigned int r = (u + 0x7FFFu + ((u >> 16) & 1u)) >> 16;
    return (unsigned short)r;
}
__device__ __forceinline__ float b2f(unsigned short h) {
    return __uint_as_float(((unsigned int)h) << 16);
}
__device__ __forceinline__ float sigmf(float x) { return 1.0f / (1.0f + __expf(-x)); }

union FragBF { v16bf v; uint4 q[2]; };

// ---------------------------------------------------------------------------
// bf16 GEMM:  C[M,N] (f32) = A[M,K] * BT[N,K]^T  (+ bias[n])
// A, BT row-major bf16 with K contiguous.
// Fast path (FULL): one wave computes a 16 x 64 tile (NT=4), K%64==0,
// M%16==0, N%64==0, software-pipelined with two fragment register sets so
// loads for stage k+1 overlap the WMMAs of stage k (partial loadcnt waits).
// Bounds path (NT=1): guarded loads/stores for the tiny M=2 decoder GEMMs.
// K multiple of 32 (fast: 64); row strides multiples of 8 elements (16B).
// ---------------------------------------------------------------------------
template<int NT>
__device__ __forceinline__ void load_frags_full(
    const unsigned short* __restrict__ Arow,
    const unsigned short* const (&Brow)[NT],
    int kb, int ashift, int bshift,
    FragBF& a, FragBF (&b)[NT])
{
    __builtin_prefetch(Arow + kb + 512, 0, 1);
    a.q[0] = *(const uint4*)(Arow + kb + ashift);        // k = kb+as .. +7
    a.q[1] = *(const uint4*)(Arow + kb + ashift + 16);   // k = kb+16+as .. +7
#pragma unroll
    for (int j = 0; j < NT; ++j) {
        b[j].q[0] = *(const uint4*)(Brow[j] + kb + bshift);      // k = kb+bs .. +7
        b[j].q[1] = *(const uint4*)(Brow[j] + kb + bshift + 16); // k = kb+16+bs .. +7
    }
}

template<int NT>
__device__ __forceinline__ void mma_frags(FragBF& a, FragBF (&b)[NT], v8f (&acc)[NT])
{
#pragma unroll
    for (int j = 0; j < NT; ++j)
        acc[j] = __builtin_amdgcn_wmma_f32_16x16x32_bf16(
                     false, a.v, false, b[j].v, (short)0, acc[j], false, false);
}

template<int NT, bool BOUNDS>
__global__ void __launch_bounds__(128)
wgemm_kernel(float* __restrict__ C,
             const unsigned short* __restrict__ A,
             const unsigned short* __restrict__ BT,
             const float* __restrict__ bias,
             int M, int N, int K, int lda, int ldbt, long long ldc,
             int mtiles, int ngroups)
{
    const int wave = threadIdx.x >> 5;
    const int lane = threadIdx.x & 31;
    const int tile = blockIdx.x * 4 + wave;
    if (tile >= mtiles * ngroups) return;   // uniform per wave: EXEC stays full

    const int mt = tile / ngroups;
    const int ng = tile % ngroups;
    const int m0 = mt * 16;
    const int n0 = ng * (16 * NT);
    const int mrow = m0 + (lane & 15);
    const int ashift = (lane >> 4) * 8;   // A lane k-offset: 0 or 8
    const int bshift = (lane >> 4) * 8;   // B lane k-offset: 0 or 8

    const unsigned short* Arow = A + (long long)mrow * lda;
    const unsigned short* Brow[NT];
#pragma unroll
    for (int j = 0; j < NT; ++j)
        Brow[j] = BT + (long long)(n0 + j * 16 + (lane & 15)) * ldbt;

    v8f acc[NT];
#pragma unroll
    for (int j = 0; j < NT; ++j) acc[j] = v8f{};

    if (!BOUNDS) {
        // -------- software-pipelined fast path: K % 64 == 0 ----------------
        FragBF a0, a1;
        FragBF b0[NT], b1[NT];
        load_frags_full<NT>(Arow, Brow, 0, ashift, bshift, a0, b0);
        for (int kb = 0; kb < K; kb += 64) {
            load_frags_full<NT>(Arow, Brow, kb + 32, ashift, bshift, a1, b1);
            mma_frags<NT>(a0, b0, acc);
            if (kb + 64 < K)
                load_frags_full<NT>(Arow, Brow, kb + 64, ashift, bshift, a0, b0);
            mma_frags<NT>(a1, b1, acc);
        }
    } else {
        // -------- guarded path (tiny decoder GEMMs) ------------------------
        bool nvalid[NT];
#pragma unroll
        for (int j = 0; j < NT; ++j) nvalid[j] = (n0 + j * 16 + (lane & 15)) < N;
        const bool mvalid = (mrow < M);

        FragBF a;
        FragBF b[NT];
        const uint4 zq = {0u, 0u, 0u, 0u};
        a.q[0] = zq; a.q[1] = zq;
#pragma unroll
        for (int j = 0; j < NT; ++j) { b[j].q[0] = zq; b[j].q[1] = zq; }

        for (int kb = 0; kb < K; kb += 32) {
            if (mvalid) {
                a.q[0] = *(const uint4*)(Arow + kb + ashift);
                a.q[1] = *(const uint4*)(Arow + kb + ashift + 16);
            }
#pragma unroll
            for (int j = 0; j < NT; ++j) {
                if (nvalid[j]) {
                    b[j].q[0] = *(const uint4*)(Brow[j] + kb + bshift);
                    b[j].q[1] = *(const uint4*)(Brow[j] + kb + bshift + 16);
                }
            }
            mma_frags<NT>(a, b, acc);
        }
    }

#pragma unroll
    for (int j = 0; j < NT; ++j) {
        const int nrow = n0 + j * 16 + (lane & 15);
        if (BOUNDS && nrow >= N) continue;
        const float bv = bias ? bias[nrow] : 0.0f;
#pragma unroll
        for (int v = 0; v < 8; ++v) {
            const int m = m0 + v + ((lane >> 4) * 8);
            if (!BOUNDS || m < M)
                C[(long long)m * ldc + nrow] = acc[j][v] + bv;
        }
    }
}

// ---------------------------------------------------------------------------
// conv1: 7x7 stride2 pad3, 1 -> 64 ch.  image [2,1,64,256] -> NHWC f32 [2,32,128,64]
// ---------------------------------------------------------------------------
__global__ void conv1_kernel(const float* __restrict__ img,
                             const float* __restrict__ w,   // [64,49]
                             const float* __restrict__ b,
                             float* __restrict__ out)
{
    long long idx = (long long)blockIdx.x * 256 + threadIdx.x;
    if (idx >= 2LL * 32 * 128 * 64) return;
    int c  = (int)(idx & 63);
    int ow = (int)((idx >> 6) & 127);
    int oh = (int)((idx >> 13) & 31);
    int bb = (int)(idx >> 18);
    float acc = b[c];
    const float* wp = w + c * 49;
    for (int ky = 0; ky < 7; ++ky) {
        int ih = oh * 2 - 3 + ky;
        if (ih < 0 || ih >= 64) continue;
        for (int kx = 0; kx < 7; ++kx) {
            int iw = ow * 2 - 3 + kx;
            if (iw < 0 || iw >= 256) continue;
            acc += img[((long long)bb * 64 + ih) * 256 + iw] * wp[ky * 7 + kx];
        }
    }
    out[idx] = acc;
}

// training-mode BN stats over rows of NHWC [M,C]: one block per channel
__global__ void bn_stats_kernel(const float* __restrict__ X, float* mean, float* invstd,
                                int M, int C)
{
    __shared__ float s1[256], s2[256];
    int c = blockIdx.x;
    float a = 0.f, b = 0.f;
    for (int i = threadIdx.x; i < M; i += 256) {
        float x = X[(long long)i * C + c];
        a += x; b += x * x;
    }
    s1[threadIdx.x] = a; s2[threadIdx.x] = b;
    __syncthreads();
    for (int off = 128; off > 0; off >>= 1) {
        if (threadIdx.x < off) { s1[threadIdx.x] += s1[threadIdx.x + off];
                                 s2[threadIdx.x] += s2[threadIdx.x + off]; }
        __syncthreads();
    }
    if (threadIdx.x == 0) {
        float m = s1[0] / M;
        float v = s2[0] / M - m * m;
        mean[c] = m;
        invstd[c] = rsqrtf(v + 1e-5f);
    }
}

// y = relu((x-mean)*invstd*g + be); optional f32 and/or bf16 outputs
__global__ void bn_act_kernel(const float* __restrict__ X,
                              float* __restrict__ outf,
                              unsigned short* __restrict__ outb,
                              const float* __restrict__ mean, const float* __restrict__ invstd,
                              const float* __restrict__ g, const float* __restrict__ be,
                              long long n, int C)
{
    long long idx = (long long)blockIdx.x * 256 + threadIdx.x;
    if (idx >= n) return;
    int c = (int)(idx % C);
    float y = (X[idx] - mean[c]) * invstd[c] * g[c] + be[c];
    y = fmaxf(y, 0.0f);
    if (outf) outf[idx] = y;
    if (outb) outb[idx] = f2b(y);
}

// maxpool 3x3 s2 pad1 over NHWC f32 [2,32,128,64] -> bf16 [2,16,64,64]
__global__ void maxpool_kernel(const float* __restrict__ in, unsigned short* __restrict__ out)
{
    long long idx = (long long)blockIdx.x * 256 + threadIdx.x;
    if (idx >= 2LL * 16 * 64 * 64) return;
    int c = (int)(idx & 63);
    int w = (int)((idx >> 6) & 63);
    int h = (int)((idx >> 12) & 15);
    int bb = (int)(idx >> 16);
    float m = -3.0e38f;
    for (int ky = 0; ky < 3; ++ky) {
        int ih = h * 2 - 1 + ky;
        if (ih < 0 || ih >= 32) continue;
        for (int kx = 0; kx < 3; ++kx) {
            int iw = w * 2 - 1 + kx;
            if (iw < 0 || iw >= 128) continue;
            m = fmaxf(m, in[(((long long)bb * 32 + ih) * 128 + iw) * 64 + c]);
        }
    }
    out[idx] = f2b(m);
}

// im2col for 3x3 pad1 conv on NHWC bf16 [2,16,64,C] -> [2048, C*9], k = c*9+tap
__global__ void im2col_kernel(const unsigned short* __restrict__ act,
                              unsigned short* __restrict__ col, int C)
{
    const int K9 = C * 9;
    long long idx = (long long)blockIdx.x * 256 + threadIdx.x;
    if (idx >= 2048LL * K9) return;
    long long m = idx / K9;
    int k = (int)(idx % K9);
    int c = k / 9, tap = k % 9;
    int w = (int)(m & 63);
    int h = (int)((m >> 6) & 15);
    int bb = (int)(m >> 10);
    int ih = h + tap / 3 - 1;
    int iw = w + tap % 3 - 1;
    unsigned short v = 0;
    if (ih >= 0 && ih < 16 && iw >= 0 && iw < 64)
        v = act[(((long long)bb * 16 + ih) * 64 + iw) * (long long)C + c];
    col[idx] = v;
}

__global__ void cvt_bf16_kernel(const float* __restrict__ src, unsigned short* __restrict__ dst,
                                long long n)
{
    long long idx = (long long)blockIdx.x * 256 + threadIdx.x;
    if (idx < n) dst[idx] = f2b(src[idx]);
}

__global__ void zero_u16_kernel(unsigned short* p, int n)
{
    int idx = blockIdx.x * 256 + threadIdx.x;
    if (idx < n) p[idx] = 0;
}

// GRU gate fuse: GI row = n*64+t (precomputed x@Wih+bih), GH [32,1536] current-step
__global__ void gru_gate_kernel(const float* __restrict__ GI, const float* __restrict__ GH,
                                unsigned short* __restrict__ h,
                                unsigned short* __restrict__ seq, int t, int dcol)
{
    int idx = blockIdx.x * 256 + threadIdx.x;
    if (idx >= 32 * 512) return;
    int n = idx >> 9, i = idx & 511;
    const float* gi = GI + ((long long)(n * 64 + t)) * 1536;
    const float* gh = GH + (long long)n * 1536;
    float r  = sigmf(gi[i]        + gh[i]);
    float z  = sigmf(gi[512 + i]  + gh[512 + i]);
    float nn = tanhf(gi[1024 + i] + r * gh[1024 + i]);
    float hp = b2f(h[idx]);
    float hn = (1.0f - z) * nn + z * hp;
    unsigned short hb = f2b(hn);
    h[idx] = hb;
    seq[((long long)(n * 64 + t)) * 1024 + dcol + i] = hb;
}

// decoder GRU gate: gi/gh [2,1536], hd bf16 [2,512] updated in place
__global__ void dec_gate_kernel(const float* __restrict__ gi, const float* __restrict__ gh,
                                unsigned short* __restrict__ hd)
{
    int idx = blockIdx.x * 256 + threadIdx.x;
    if (idx >= 2 * 512) return;
    int n = idx >> 9, i = idx & 511;
    const float* a = gi + (long long)n * 1536;
    const float* b = gh + (long long)n * 1536;
    float r  = sigmf(a[i]        + b[i]);
    float z  = sigmf(a[512 + i]  + b[512 + i]);
    float nn = tanhf(a[1024 + i] + r * b[1024 + i]);
    float hp = b2f(hd[idx]);
    hd[idx] = f2b((1.0f - z) * nn + z * hp);
}

// embed lookup -> xin[:, 0:256] bf16; tokens = [BOS=1, target[:, :-1]]
__global__ void emb_kernel(const float* __restrict__ embed, const int* __restrict__ target,
                           unsigned short* __restrict__ xin, int t)
{
    int idx = blockIdx.x * 256 + threadIdx.x;
    if (idx >= 2 * 256) return;
    int b = idx >> 8, i = idx & 255;
    int tok = (t == 0) ? 1 : target[b * 64 + (t - 1)];
    if (tok < 0) tok = 0; if (tok >= 600) tok = 599;
    xin[b * 1280 + i] = f2b(embed[(long long)tok * 256 + i]);
}

// Bahdanau attention with precomputed encW = enc@We+b: score/softmax/context
// hidW [2,512], encW [2048,512], encoded bf16 [2048,1024]; ctx -> xin[:,256:1280]
__global__ void __launch_bounds__(256)
attn_ctx_kernel(const float* __restrict__ hidW, const float* __restrict__ encW,
                const float* __restrict__ attn_v, const unsigned short* __restrict__ enc,
                unsigned short* __restrict__ xin)
{
    __shared__ float sc[1024];
    __shared__ float red[256];
    const int b = blockIdx.x, tid = threadIdx.x;

    for (int s = tid; s < 1024; s += 256) {
        const float* ew = encW + ((long long)(b * 1024 + s)) * 512;
        const float* hw = hidW + b * 512;
        float dot = 0.0f;
        for (int i = 0; i < 512; ++i)
            dot += tanhf(hw[i] + ew[i]) * attn_v[i];
        sc[s] = dot;
    }
    __syncthreads();

    float m = -3.0e38f;
    for (int s = tid; s < 1024; s += 256) m = fmaxf(m, sc[s]);
    red[tid] = m; __syncthreads();
    for (int off = 128; off > 0; off >>= 1) {
        if (tid < off) red[tid] = fmaxf(red[tid], red[tid + off]);
        __syncthreads();
    }
    const float gmax = red[0];
    __syncthreads();

    float sum = 0.0f;
    for (int s = tid; s < 1024; s += 256) {
        float e = __expf(sc[s] - gmax);
        sc[s] = e;
        sum += e;
    }
    red[tid] = sum; __syncthreads();
    for (int off = 128; off > 0; off >>= 1) {
        if (tid < off) red[tid] += red[tid + off];
        __syncthreads();
    }
    const float inv = 1.0f / red[0];
    __syncthreads();

    for (int e = tid; e < 1024; e += 256) {
        float acc = 0.0f;
        const unsigned short* ep = enc + (long long)b * 1024 * 1024 + e;
        for (int s = 0; s < 1024; ++s)
            acc += sc[s] * b2f(ep[(long long)s * 1024]);
        xin[b * 1280 + 256 + e] = f2b(acc * inv);
    }
}

// ---------------------------------------------------------------------------
// Host side
// ---------------------------------------------------------------------------
struct NetP {
    const float *image;
    const int   *target;
    const float *conv1_w, *conv1_b, *bn1_g, *bn1_b;
    const float *lw[13], *lb[13], *lg[13], *lbe[13];
    const float *e_wih[4], *e_whh[4], *e_bih[4], *e_bhh[4];  // l0f,l0b,l1f,l1b
    const float *embed, *attn_w, *attn_b, *attn_v;
    const float *dec_w_ih, *dec_w_hh, *dec_b_ih, *dec_b_hh, *fc_w, *fc_b;
};

static void map_inputs(void* const* d_in, const int* in_sizes, int n_in, NetP& p)
{
    const bool ins = (n_in >= 2 && in_sizes[1] == 128);  // target second => insertion order
    int i = 0;
    auto F = [&]() { return (const float*)d_in[i++]; };
    const int counts[3] = {3, 4, 6};
    if (ins) {
        p.image = F(); p.target = (const int*)d_in[i++];
        p.conv1_w = F(); p.conv1_b = F(); p.bn1_g = F(); p.bn1_b = F();
        int bi = 0;
        for (int L = 0; L < 3; ++L)
            for (int k = 0; k < counts[L]; ++k) {
                p.lw[bi] = F(); p.lb[bi] = F(); p.lg[bi] = F(); p.lbe[bi] = F(); ++bi;
            }
        for (int e = 0; e < 4; ++e) { p.e_wih[e] = F(); p.e_whh[e] = F(); p.e_bih[e] = F(); p.e_bhh[e] = F(); }
        p.embed = F(); p.attn_w = F(); p.attn_b = F(); p.attn_v = F();
        p.dec_w_ih = F(); p.dec_w_hh = F(); p.dec_b_ih = F(); p.dec_b_hh = F();
        p.fc_w = F(); p.fc_b = F();
    } else {  // pytree-sorted keys; target last
        p.image = F();
        p.attn_b = F(); p.attn_v = F(); p.attn_w = F();
        p.bn1_b = F(); p.bn1_g = F();
        p.conv1_b = F(); p.conv1_w = F();
        p.dec_b_hh = F(); p.dec_b_ih = F(); p.dec_w_hh = F(); p.dec_w_ih = F();
        p.embed = F();
        for (int e = 0; e < 4; ++e) { p.e_bhh[e] = F(); p.e_bih[e] = F(); p.e_whh[e] = F(); p.e_wih[e] = F(); }
        p.fc_b = F(); p.fc_w = F();
        int bi = 0;
        for (int L = 0; L < 3; ++L)
            for (int k = 0; k < counts[L]; ++k) {
                p.lb[bi] = F(); p.lbe[bi] = F(); p.lg[bi] = F(); p.lw[bi] = F(); ++bi;
            }
        p.target = (const int*)d_in[i++];
    }
}

// Full tiles (M%16==0, N%64==0, K%64==0): pipelined 16x64 per wave.
// Otherwise: 16x16 per wave with M/N guards (tiny decoder GEMMs only).
static inline void gemm(hipStream_t s, float* C, const unsigned short* A,
                        const unsigned short* BT, const float* bias,
                        int M, int N, int K, int lda, int ldbt, long long ldc)
{
    if ((M % 16 == 0) && (N % 64 == 0) && (K % 64 == 0)) {
        int mt = M / 16, ng = N / 64;
        int blocks = (mt * ng + 3) / 4;
        wgemm_kernel<4, false><<<blocks, 128, 0, s>>>(C, A, BT, bias, M, N, K,
                                                      lda, ldbt, ldc, mt, ng);
    } else {
        int mt = (M + 15) / 16, ng = (N + 15) / 16;
        int blocks = (mt * ng + 3) / 4;
        wgemm_kernel<1, true><<<blocks, 128, 0, s>>>(C, A, BT, bias, M, N, K,
                                                     lda, ldbt, ldc, mt, ng);
    }
}

static inline void cvt(hipStream_t s, const float* src, unsigned short* dst, long long n)
{
    cvt_bf16_kernel<<<(int)((n + 255) / 256), 256, 0, s>>>(src, dst, n);
}

extern "C" void kernel_launch(void* const* d_in, const int* in_sizes, int n_in,
                              void* d_out, int out_size, void* d_ws, size_t ws_size,
                              hipStream_t stream)
{
    NetP P;
    map_inputs(d_in, in_sizes, n_in, P);

    const long long MB = 1024LL * 1024LL;
    char* ws = (char*)d_ws;
    float*          cf32   = (float*)(ws + 0 * MB);        // up to 4 MB GEMM/conv f32 out
    float*          normf  = (float*)(ws + 4 * MB);        // conv1 normalized f32 (2 MB)
    unsigned short* act0   = (unsigned short*)(ws + 6 * MB);
    unsigned short* act1   = (unsigned short*)(ws + 8 * MB);
    void*           big    = (void*)(ws + 10 * MB);        // im2col bf16 (19MB) / GI f32 (12.6MB)
    unsigned short* wconv  = (unsigned short*)(ws + 30 * MB);
    float*          encW   = (float*)(ws + 35 * MB);       // 4 MB
    float*          meanb  = (float*)(ws + 39 * MB);
    float*          istdb  = meanb + 512;
    float*          GH     = (float*)(ws + 40 * MB);       // [32,1536]
    unsigned short* hbuf   = (unsigned short*)(ws + 41 * MB);
    unsigned short* seq0   = (unsigned short*)(ws + 42 * MB);  // 4 MB
    unsigned short* seq1   = (unsigned short*)(ws + 46 * MB);  // 4 MB  (== encoded)
    unsigned short* wenc   = (unsigned short*)(ws + 50 * MB);  // ~15.7 MB
    unsigned short* wattn  = (unsigned short*)(ws + 66 * MB);
    unsigned short* wdecih = (unsigned short*)(ws + 68 * MB);
    unsigned short* wdechh = (unsigned short*)(ws + 72 * MB);
    unsigned short* wfc    = (unsigned short*)(ws + 74 * MB);
    unsigned short* hd     = (unsigned short*)(ws + 75 * MB);
    unsigned short* xin    = (unsigned short*)(ws + 75 * MB + 8192);
    float*          gidec  = (float*)(ws + 75 * MB + 32768);
    float*          ghdec  = (float*)(ws + 75 * MB + 65536);
    float*          hidW   = (float*)(ws + 75 * MB + 98304);
    (void)ws_size;

    // ---------------- Phase 1: stem conv + BN + ReLU + maxpool ----------------
    {
        long long n = 2LL * 32 * 128 * 64;
        conv1_kernel<<<(int)((n + 255) / 256), 256, 0, stream>>>(P.image, P.conv1_w, P.conv1_b, cf32);
        bn_stats_kernel<<<64, 256, 0, stream>>>(cf32, meanb, istdb, 8192, 64);
        bn_act_kernel<<<(int)((n + 255) / 256), 256, 0, stream>>>(
            cf32, normf, nullptr, meanb, istdb, P.bn1_g, P.bn1_b, n, 64);
        long long np = 2LL * 16 * 64 * 64;
        maxpool_kernel<<<(int)((np + 255) / 256), 256, 0, stream>>>(normf, act0);
    }

    // ---------------- Phase 2: 13 conv3x3 + BN + ReLU blocks (WMMA GEMM) ----
    unsigned short* feat;
    {
        const int cout_cfg[13] = {128,128,128, 256,256,256,256, 512,512,512,512,512,512};
        int cin = 64;
        unsigned short* ain = act0;
        unsigned short* aout = act1;
        for (int bi = 0; bi < 13; ++bi) {
            int cout = cout_cfg[bi];
            int K9 = cin * 9;
            cvt(stream, P.lw[bi], wconv, (long long)cout * K9);
            long long coln = 2048LL * K9;
            im2col_kernel<<<(int)((coln + 255) / 256), 256, 0, stream>>>(ain, (unsigned short*)big, cin);
            gemm(stream, cf32, (unsigned short*)big, wconv, P.lb[bi],
                 2048, cout, K9, K9, K9, cout);
            bn_stats_kernel<<<cout, 256, 0, stream>>>(cf32, meanb, istdb, 2048, cout);
            long long n = 2048LL * cout;
            bn_act_kernel<<<(int)((n + 255) / 256), 256, 0, stream>>>(
                cf32, nullptr, aout, meanb, istdb, P.lg[bi], P.lbe[bi], n, cout);
            unsigned short* t = ain; ain = aout; aout = t;
            cin = cout;
        }
        feat = ain;  // final activations [2048, 512] bf16
    }

    // ---------------- Phase 3: 2-layer bidirectional GRU encoder ------------
    unsigned short* wih_bf[4];
    unsigned short* whh_bf[4];
    {
        long long off = 0;
        const int din[2] = {512, 1024};
        for (int l = 0; l < 2; ++l)
            for (int d = 0; d < 2; ++d) {
                int e = l * 2 + d;
                wih_bf[e] = wenc + off;
                cvt(stream, P.e_wih[e], wih_bf[e], 1536LL * din[l]);
                off += 1536LL * din[l];
                whh_bf[e] = wenc + off;
                cvt(stream, P.e_whh[e], whh_bf[e], 1536LL * 512);
                off += 1536LL * 512;
            }
    }
    {
        float* GI = (float*)big;
        const unsigned short* layer_in = feat;
        int K = 512;
        unsigned short* seqout = seq0;
        for (int l = 0; l < 2; ++l) {
            for (int d = 0; d < 2; ++d) {
                int e = l * 2 + d;
                // GI = X @ Wih^T + bih for all (n,t): [2048,1536]
                gemm(stream, GI, layer_in, wih_bf[e], P.e_bih[e],
                     2048, 1536, K, K, K, 1536);
                zero_u16_kernel<<<(32 * 512 + 255) / 256, 256, 0, stream>>>(hbuf, 32 * 512);
                for (int step = 0; step < 64; ++step) {
                    int t = (d == 0) ? step : (63 - step);
                    gemm(stream, GH, hbuf, whh_bf[e], P.e_bhh[e],
                         32, 1536, 512, 512, 512, 1536);
                    gru_gate_kernel<<<64, 256, 0, stream>>>(GI, GH, hbuf, seqout, t, d * 512);
                }
            }
            layer_in = seqout;
            K = 1024;
            seqout = seq1;
        }
    }
    // seq1 == encoded, bf16 [2048, 1024], row m = b*1024 + s

    // ---------------- Phase 4: attention decoder ----------------------------
    cvt(stream, P.attn_w,   wattn,  512LL * 1536);
    cvt(stream, P.dec_w_ih, wdecih, 1536LL * 1280);
    cvt(stream, P.dec_w_hh, wdechh, 1536LL * 512);
    cvt(stream, P.fc_w,     wfc,    600LL * 512);

    // Hoisted: encW = encoded @ We^T + attn_b   (We = attn_w[:, 512:1536])
    gemm(stream, encW, seq1, wattn + 512, P.attn_b,
         2048, 512, 1024, 1024, 1536, 512);

    zero_u16_kernel<<<(2 * 512 + 255) / 256, 256, 0, stream>>>(hd, 2 * 512);

    float* out = (float*)d_out;
    (void)out_size;
    for (int t = 0; t < 64; ++t) {
        // hidW = hd @ Wh^T   (Wh = attn_w[:, :512]); bias already in encW
        gemm(stream, hidW, hd, wattn, nullptr, 2, 512, 512, 512, 1536, 512);
        emb_kernel<<<2, 256, 0, stream>>>(P.embed, P.target, xin, t);
        attn_ctx_kernel<<<2, 256, 0, stream>>>(hidW, encW, P.attn_v, seq1, xin);
        // gi = [emb,ctx] @ dec_w_ih^T + b_ih ; gh = hd @ dec_w_hh^T + b_hh
        gemm(stream, gidec, xin, wdecih, P.dec_b_ih, 2, 1536, 1280, 1280, 1280, 1536);
        gemm(stream, ghdec, hd,  wdechh, P.dec_b_hh, 2, 1536, 512, 512, 512, 1536);
        dec_gate_kernel<<<4, 256, 0, stream>>>(gidec, ghdec, hd);
        // logits[b, t, :] = hn @ fc_w^T + fc_b   (strided C: ldc = T*VOCAB)
        gemm(stream, out + (long long)t * 600, hd, wfc, P.fc_b,
             2, 600, 512, 512, 512, 64LL * 600);
    }
}
